// SIIE_torch_82403242541482
// MI455X (gfx1250) — compile-verified, matrix operand order checked by reading, not run on hardware
//
#include <hip/hip_runtime.h>
#include <math.h>

typedef __attribute__((ext_vector_type(16))) _Float16 v16h;
typedef __attribute__((ext_vector_type(8)))  float    v8f;

#define NB     16
#define INSZ   150
#define NPIX   (INSZ*INSZ)      // 22500
#define HB     61
#define HP     64               // padded histogram edge
#define FEPS   1e-6f
#define SLICES 16               // K-dim split of the histogram GEMM

// ---- ISA 7.12.2: 16-bit A-matrix 16x32 per-lane K index for half i (0..15) ----
__device__ __forceinline__ int kmap16(int lane, int i) {
  int base = (lane & 16) ? 8 : 0;
  int jv = i >> 1, pos = i & 1;
  return (jv < 4) ? (base + 2*jv + pos) : (16 + base + 2*(jv - 4) + pos);
}

// =====================================================================
// 1) RGB-uv soft histogram as a WMMA GEMM, K-split across SLICES:
//    hist[u,v] = sum_p exp(-(u_p-bin_u)^2/su^2)*w_p * exp(-(v_p-bin_v)^2/sv^2)
//    grid = (SLICES, NB*3); block = 4 waves. Each wave owns a 64x64 f32
//    accumulator (16 WMMA tiles) over disjoint 32-pixel chunks; the 4 waves
//    reduce in LDS and the block writes its partial 61x61 tile to `part`.
//    Deterministic: fixed chunk->wave assignment, fixed slice-sum order later.
// =====================================================================
__global__ __launch_bounds__(128)
void hist_kernel(const float* __restrict__ img, const float* __restrict__ su,
                 const float* __restrict__ sv, const float* __restrict__ cw,
                 float* __restrict__ part)
{
  __shared__ float acc_s[HP * HP];            // 16 KB reduction tile
  __shared__ float us[4][32], vs[4][32], wt[4][32];

  const int slice = blockIdx.x;
  const int bc = blockIdx.y;
  const int b = bc / 3, c = bc % 3;
  const int tid = threadIdx.x;
  const int wave = tid >> 5, lane = tid & 31;

  const float inv_su2 = 1.0f / (su[c] * su[c]);
  const float inv_sv2 = 1.0f / (sv[c] * sv[c]);
  const float cweight = cw[c];

  const float* pr = img + (size_t)(b*3 + 0) * NPIX;
  const float* pg = img + (size_t)(b*3 + 1) * NPIX;
  const float* pb = img + (size_t)(b*3 + 2) * NPIX;

  v8f acc[4][4];
  #pragma unroll
  for (int tm = 0; tm < 4; ++tm)
    #pragma unroll
    for (int tn = 0; tn < 4; ++tn)
      acc[tm][tn] = (v8f){0.f,0.f,0.f,0.f,0.f,0.f,0.f,0.f};

  const int nchunks = (NPIX + 31) / 32;       // 704 = 64 * 11 -> uniform trip count
  const int ml = lane & 15;

  for (int ch = slice * 4 + wave; ch < nchunks; ch += 4 * SLICES) {
    int p = ch * 32 + lane;
    float r = 0.f, g = 0.f, bl = 0.f, w = 0.f;
    if (p < NPIX) {
      r  = fminf(fmaxf(pr[p], 0.f), 1.f);
      g  = fminf(fmaxf(pg[p], 0.f), 1.f);
      bl = fminf(fmaxf(pb[p], 0.f), 1.f);
      w  = sqrtf(r*r + g*g + bl*bl) * cweight;
    }
    float lr = __logf(r + FEPS), lg = __logf(g + FEPS), lb = __logf(bl + FEPS);
    float u, v;
    if      (c == 0) { u = lr - lg; v = lr - lb; }
    else if (c == 1) { u = lg - lr; v = lg - lb; }
    else             { u = lb - lr; v = lb - lg; }

    __syncthreads();                           // WAR vs previous iter reads
    us[wave][lane] = u;
    vs[wave][lane] = v;
    wt[wave][lane] = (p < NPIX) ? w : 0.f;
    __syncthreads();

    v16h afr[4], bfr[4];
    #pragma unroll
    for (int t = 0; t < 4; ++t) {
      int   m   = t*16 + ml;
      float bm  = -3.f + 0.1f * (float)m;
      bool  okm = (m < HB);
      #pragma unroll
      for (int i = 0; i < 16; ++i) {
        int k = kmap16(lane, i);
        float du = us[wave][k] - bm;
        float av = okm ? __expf(-du*du*inv_su2) * wt[wave][k] : 0.f;
        afr[t][i] = (_Float16)av;
        float dv = vs[wave][k] - bm;
        float bv = okm ? __expf(-dv*dv*inv_sv2) : 0.f;
        bfr[t][i] = (_Float16)bv;
      }
    }
    #pragma unroll
    for (int tm = 0; tm < 4; ++tm)
      #pragma unroll
      for (int tn = 0; tn < 4; ++tn)
        acc[tm][tn] = __builtin_amdgcn_wmma_f32_16x16x32_f16(
            false, afr[tm], false, bfr[tn], (short)0, acc[tm][tn], false, false);
  }

  // cross-wave reduction in LDS (ds_add_f32)
  for (int i = tid; i < HP*HP; i += 128) acc_s[i] = 0.f;
  __syncthreads();
  int n_l = lane & 15, m_off = (lane >> 4) * 8;
  #pragma unroll
  for (int tm = 0; tm < 4; ++tm)
    #pragma unroll
    for (int tn = 0; tn < 4; ++tn)
      #pragma unroll
      for (int r = 0; r < 8; ++r)
        atomicAdd(&acc_s[(tm*16 + m_off + r) * HP + tn*16 + n_l], acc[tm][tn][r]);
  __syncthreads();

  float* hout = part + ((size_t)slice * (NB*3) + bc) * (HB*HB);
  for (int i = tid; i < HB*HB; i += 128) {
    int rr = i / HB, cc = i % HB;
    hout[i] = acc_s[rr * HP + cc];
  }
}

// ---------------------------------------------------------------------
// 2) slice-sum + per-batch normalization:
//    hist[b] = sum_s part[s][b]; hist[b] /= (sum over 3*61*61 + eps)
// ---------------------------------------------------------------------
__global__ __launch_bounds__(256)
void hist_norm_kernel(const float* __restrict__ part, float* __restrict__ hist)
{
  __shared__ float red[256];
  const int b = blockIdx.x;
  const int n = 3 * HB * HB;
  float* h = hist + (size_t)b * n;
  float tot = 0.f;
  for (int i = threadIdx.x; i < n; i += 256) {
    int c = i / (HB*HB), j = i % (HB*HB);
    float s = 0.f;
    #pragma unroll
    for (int sl = 0; sl < SLICES; ++sl)
      s += part[((size_t)sl * (NB*3) + b*3 + c) * (HB*HB) + j];
    h[i] = s;
    tot += s;
  }
  red[threadIdx.x] = tot; __syncthreads();
  for (int o = 128; o > 0; o >>= 1) {
    if (threadIdx.x < o) red[threadIdx.x] += red[threadIdx.x + o];
    __syncthreads();
  }
  float inv = 1.f / (red[0] + FEPS);
  for (int i = threadIdx.x; i < n; i += 256) h[i] *= inv;
}

// ---------------------------------------------------------------------
// 3) implicit-GEMM conv + bias + relu via WMMA f16->f32.
//    M = Cout, N = Ho*Wo, K = Cin*KH*KW. One wave per 16x16 output tile.
// ---------------------------------------------------------------------
__global__ __launch_bounds__(128)
void conv_wmma_kernel(const float* __restrict__ in, const float* __restrict__ wgt,
                      const float* __restrict__ bias, float* __restrict__ out,
                      int Cin, int H, int W, int Cout, int KH, int KW,
                      int stride, int Ho, int Wo)
{
  const int b    = blockIdx.z;
  const int mt   = blockIdx.x;
  const int lane = threadIdx.x & 31;
  const int wave = threadIdx.x >> 5;
  const int nt   = blockIdx.y * 4 + wave;
  const int N    = Ho * Wo;
  if (nt * 16 >= N) return;                    // whole wave exits (EXEC stays full)

  const int K     = Cin * KH * KW;
  const int KHW   = KH * KW;
  const int mBase = mt * 16, nBase = nt * 16;
  const int ml    = lane & 15;
  const int m     = mBase + ml;
  const int n     = nBase + ml;
  const int oy    = n / Wo, ox = n % Wo;
  const bool okn  = (n < N);
  const bool okm  = (m < Cout);
  const float* inb = in + (size_t)b * Cin * H * W;

  v8f acc = (v8f){0.f,0.f,0.f,0.f,0.f,0.f,0.f,0.f};

  for (int kc = 0; kc < K; kc += 32) {
    v16h afr, bfr;
    #pragma unroll
    for (int i = 0; i < 16; ++i) {
      int k = kc + kmap16(lane, i);
      float av = (k < K && okm) ? wgt[(size_t)m * K + k] : 0.f;
      afr[i] = (_Float16)av;
      float bv = 0.f;
      if (k < K && okn) {
        int cin = k / KHW, rem = k % KHW;
        int ky = rem / KW, kx = rem % KW;
        bv = inb[((size_t)cin * H + (oy * stride + ky)) * W + (ox * stride + kx)];
      }
      bfr[i] = (_Float16)bv;
    }
    acc = __builtin_amdgcn_wmma_f32_16x16x32_f16(
        false, afr, false, bfr, (short)0, acc, false, false);
  }

  const int n_l = lane & 15, m_off = (lane >> 4) * 8;
  const int nn  = nBase + n_l;
  #pragma unroll
  for (int r = 0; r < 8; ++r) {
    int mm = mBase + m_off + r;
    if (mm < Cout && nn < N) {
      float v = acc[r] + bias[mm];
      out[((size_t)b * Cout + mm) * N + nn] = fmaxf(v, 0.f);
    }
  }
}

// ---------------------------------------------------------------------
// 4) |z @ fc^T| : one block per (b, j), K = 86528
// ---------------------------------------------------------------------
__global__ __launch_bounds__(256)
void fc_abs_kernel(const float* __restrict__ z, const float* __restrict__ fc,
                   float* __restrict__ out, int M, int K)
{
  __shared__ float red[256];
  const int b = blockIdx.x / M, j = blockIdx.x % M;
  const float* zb = z  + (size_t)b * K;
  const float* fj = fc + (size_t)j * K;
  float s = 0.f;
  for (int i = threadIdx.x; i < K; i += 256) s += zb[i] * fj[i];
  red[threadIdx.x] = s; __syncthreads();
  for (int o = 128; o > 0; o >>= 1) {
    if (threadIdx.x < o) red[threadIdx.x] += red[threadIdx.x + o];
    __syncthreads();
  }
  if (threadIdx.x == 0) out[b * M + j] = fabsf(red[0]);
}

// ---------------------------------------------------------------------
// 5) build m (transpose, L1-max-row normalize) and m^-1 per batch
// ---------------------------------------------------------------------
__global__ __launch_bounds__(32)
void build_m_kernel(const float* __restrict__ head,
                    float* __restrict__ m_out, float* __restrict__ minv_out)
{
  int b = threadIdx.x;
  if (b >= NB) return;
  const float* h = head + b * 9;
  float m[3][3];
  #pragma unroll
  for (int r = 0; r < 3; ++r)
    #pragma unroll
    for (int c = 0; c < 3; ++c)
      m[r][c] = h[3*c + r];                     // reshape(3,3) then transpose
  float nmax = 0.f;
  #pragma unroll
  for (int r = 0; r < 3; ++r)
    nmax = fmaxf(nmax, fabsf(m[r][0]) + fabsf(m[r][1]) + fabsf(m[r][2]));
  float inv_n = 1.f / (nmax + 1e-4f);
  #pragma unroll
  for (int r = 0; r < 3; ++r)
    #pragma unroll
    for (int c = 0; c < 3; ++c)
      m[r][c] *= inv_n;

  float det = m[0][0]*(m[1][1]*m[2][2] - m[1][2]*m[2][1])
            - m[0][1]*(m[1][0]*m[2][2] - m[1][2]*m[2][0])
            + m[0][2]*(m[1][0]*m[2][1] - m[1][1]*m[2][0]);
  float id = 1.f / det;
  float inv[3][3];
  inv[0][0] = (m[1][1]*m[2][2] - m[1][2]*m[2][1]) * id;
  inv[0][1] = (m[0][2]*m[2][1] - m[0][1]*m[2][2]) * id;
  inv[0][2] = (m[0][1]*m[1][2] - m[0][2]*m[1][1]) * id;
  inv[1][0] = (m[1][2]*m[2][0] - m[1][0]*m[2][2]) * id;
  inv[1][1] = (m[0][0]*m[2][2] - m[0][2]*m[2][0]) * id;
  inv[1][2] = (m[0][2]*m[1][0] - m[0][0]*m[1][2]) * id;
  inv[2][0] = (m[1][0]*m[2][1] - m[1][1]*m[2][0]) * id;
  inv[2][1] = (m[0][1]*m[2][0] - m[0][0]*m[2][1]) * id;
  inv[2][2] = (m[0][0]*m[1][1] - m[0][1]*m[1][0]) * id;

  #pragma unroll
  for (int r = 0; r < 3; ++r)
    #pragma unroll
    for (int c = 0; c < 3; ++c) {
      m_out[b*9 + r*3 + c]    = m[r][c];
      minv_out[b*9 + r*3 + c] = inv[r][c];
    }
}

// ---------------------------------------------------------------------
// 6) mapped[b,i,:] = sum_j m[b,i,j] * image[b,j,:]
// ---------------------------------------------------------------------
__global__ __launch_bounds__(256)
void map_kernel(const float* __restrict__ img, const float* __restrict__ m,
                float* __restrict__ mapped)
{
  int idx = blockIdx.x * 256 + threadIdx.x;
  const int total = NB * 3 * NPIX;
  if (idx >= total) return;
  int p  = idx % NPIX;
  int bi = idx / NPIX;
  int i  = bi % 3, b = bi / 3;
  const float* mb = m + b * 9;
  float v = mb[i*3+0] * img[(size_t)(b*3+0)*NPIX + p]
          + mb[i*3+1] * img[(size_t)(b*3+1)*NPIX + p]
          + mb[i*3+2] * img[(size_t)(b*3+2)*NPIX + p];
  mapped[idx] = v;
}

// ---------------------------------------------------------------------
// 7) est_ill[b,i] = sum_j minv[b,i,j] * ill[b,j]
// ---------------------------------------------------------------------
__global__ __launch_bounds__(64)
void final_kernel(const float* __restrict__ minv, const float* __restrict__ ill,
                  float* __restrict__ out)
{
  int t = threadIdx.x;
  if (t >= NB * 3) return;
  int b = t / 3, i = t % 3;
  const float* mv = minv + b * 9;
  const float* il = ill  + b * 3;
  out[t] = mv[i*3+0]*il[0] + mv[i*3+1]*il[1] + mv[i*3+2]*il[2];
}

// =====================================================================
extern "C" void kernel_launch(void* const* d_in, const int* in_sizes, int n_in,
                              void* d_out, int out_size, void* d_ws, size_t ws_size,
                              hipStream_t stream)
{
  (void)in_sizes; (void)n_in; (void)out_size; (void)ws_size;

  const float* image = (const float*)d_in[0];
  const float* su_s  = (const float*)d_in[1];
  const float* sv_s  = (const float*)d_in[2];
  const float* c_s   = (const float*)d_in[3];
  const float* w1_s  = (const float*)d_in[4];
  const float* b1_s  = (const float*)d_in[5];
  const float* w2_s  = (const float*)d_in[6];
  const float* b2_s  = (const float*)d_in[7];
  const float* w3_s  = (const float*)d_in[8];
  const float* b3_s  = (const float*)d_in[9];
  const float* fc_s  = (const float*)d_in[10];
  const float* su_i  = (const float*)d_in[11];
  const float* sv_i  = (const float*)d_in[12];
  const float* c_i   = (const float*)d_in[13];
  const float* w1_i  = (const float*)d_in[14];
  const float* b1_i  = (const float*)d_in[15];
  const float* w2_i  = (const float*)d_in[16];
  const float* b2_i  = (const float*)d_in[17];
  const float* w3_i  = (const float*)d_in[18];
  const float* b3_i  = (const float*)d_in[19];
  const float* fc_i  = (const float*)d_in[20];
  float* out = (float*)d_out;

  // workspace layout (float offsets)
  float* ws = (float*)d_ws;
  size_t off = 0;
  float* part   = ws + off; off += (size_t)SLICES * NB * 3 * HB * HB;  // 2857728
  float* hist   = ws + off; off += (size_t)NB * 3 * HB * HB;           // 178608
  float* z1     = ws + off; off += (size_t)NB * 128 * 29 * 29;         // 1722368
  float* z2     = ws + off; off += (size_t)NB * 256 * 14 * 14;         // 802816
  float* z3     = ws + off; off += (size_t)NB * 512 * 13 * 13;         // 1384448
  float* head   = ws + off; off += (size_t)NB * 9;
  float* mmat   = ws + off; off += (size_t)NB * 9;
  float* minv   = ws + off; off += (size_t)NB * 9;
  float* mapped = ws + off; off += (size_t)NB * 3 * NPIX;              // 1080000
  float* ill    = ws + off; off += (size_t)NB * 3;

  const int FCK = 512 * 13 * 13;   // 86528

  // ---------- sensor branch ----------
  hist_kernel<<<dim3(SLICES, NB*3), 128, 0, stream>>>(image, su_s, sv_s, c_s, part);
  hist_norm_kernel<<<NB, 256, 0, stream>>>(part, hist);

  conv_wmma_kernel<<<dim3(8, 14, NB), 128, 0, stream>>>(
      hist, w1_s, b1_s, z1, 3, 61, 61, 128, 5, 5, 2, 29, 29);
  conv_wmma_kernel<<<dim3(16, 4, NB), 128, 0, stream>>>(
      z1, w2_s, b2_s, z2, 128, 29, 29, 256, 3, 3, 2, 14, 14);
  conv_wmma_kernel<<<dim3(32, 3, NB), 128, 0, stream>>>(
      z2, w3_s, b3_s, z3, 256, 14, 14, 512, 2, 2, 1, 13, 13);

  fc_abs_kernel<<<NB * 9, 256, 0, stream>>>(z3, fc_s, head, 9, FCK);
  build_m_kernel<<<1, 32, 0, stream>>>(head, mmat, minv);
  map_kernel<<<(NB*3*NPIX + 255) / 256, 256, 0, stream>>>(image, mmat, mapped);

  // ---------- illuminant branch ----------
  hist_kernel<<<dim3(SLICES, NB*3), 128, 0, stream>>>(mapped, su_i, sv_i, c_i, part);
  hist_norm_kernel<<<NB, 256, 0, stream>>>(part, hist);

  conv_wmma_kernel<<<dim3(8, 14, NB), 128, 0, stream>>>(
      hist, w1_i, b1_i, z1, 3, 61, 61, 128, 5, 5, 2, 29, 29);
  conv_wmma_kernel<<<dim3(16, 4, NB), 128, 0, stream>>>(
      z1, w2_i, b2_i, z2, 128, 29, 29, 256, 3, 3, 2, 14, 14);
  conv_wmma_kernel<<<dim3(32, 3, NB), 128, 0, stream>>>(
      z2, w3_i, b3_i, z3, 256, 14, 14, 512, 2, 2, 1, 13, 13);

  fc_abs_kernel<<<NB * 3, 256, 0, stream>>>(z3, fc_i, ill, 3, FCK);
  final_kernel<<<1, 64, 0, stream>>>(minv, ill, out);
}
